// TransformerBlock_85564338471570
// MI455X (gfx1250) — matrix operand order repaired and checked by reading, not compiled, not run on hardware
//
#include <hip/hip_runtime.h>
#include <hip/hip_bf16.h>
#include <math.h>

// ---------------------------------------------------------------------------
// MI455X (gfx1250) transformer block: bf16 WMMA with fp32 accumulation,
// Tensor Data Mover (TDM) staging of bf16 activation tiles into LDS,
// double-buffered K loop so TDM + weight VMEM overlap the WMMA stream.
// Compute-bound: ~206 GFLOP vs ~100MB unique bytes (fits in 192MB L2).
// ---------------------------------------------------------------------------

typedef __bf16 bf16;
typedef __attribute__((ext_vector_type(16))) __bf16 v16bf;
typedef __attribute__((ext_vector_type(8)))  __bf16 v8bf;
typedef __attribute__((ext_vector_type(4)))  __bf16 v4bf;
typedef __attribute__((ext_vector_type(8)))  float  v8f;
typedef __attribute__((ext_vector_type(4)))  unsigned v4u;
typedef __attribute__((ext_vector_type(8)))  int    v8i;
typedef __attribute__((ext_vector_type(4)))  int    v4i;

#define T_TOK  4096
#define DMODEL 1024
#define NHEAD  16
#define HDIM   64
#define FFDIM  4096

static __device__ inline bf16 to_bf16(float f) {
  unsigned u = __builtin_bit_cast(unsigned, f);
  u += 0x7FFFu + ((u >> 16) & 1u);                 // round-to-nearest-even
  unsigned short h = (unsigned short)(u >> 16);
  return __builtin_bit_cast(bf16, h);
}

static __device__ inline v16bf cat8(v8bf lo, v8bf hi) {
  return __builtin_shufflevector(lo, hi, 0,1,2,3,4,5,6,7,8,9,10,11,12,13,14,15);
}

static __device__ inline v8f wmma_bf16(v16bf a, v16bf b, v8f c) {
  return __builtin_amdgcn_wmma_f32_16x16x32_bf16(false, a, false, b,
                                                 (short)0, c, false, false);
}

static __device__ inline v8f v8f_zero() {
  v8f z = {0.f,0.f,0.f,0.f,0.f,0.f,0.f,0.f};
  return z;
}

// ---------------------------------------------------------------------------
// TDM: DMA a [tile_h x tile_w] bf16 tile (row stride = row_stride_elems) from
// global memory into LDS at lds_addr. D# per CDNA5 ISA 8.3/8.4 (2D tile:
// groups 2/3 zero). Caller: issue from ONE wave, then s_wait_tensorcnt +
// workgroup barrier before any wave reads the LDS tile.
// ---------------------------------------------------------------------------
static __device__ inline void tdm_load_tile_bf16(const bf16* g_tile,
                                                 unsigned lds_addr,
                                                 unsigned tile_w, unsigned tile_h,
                                                 unsigned row_stride_elems) {
  const unsigned long long ga = (unsigned long long)(uintptr_t)g_tile;
  v4u g0 = { 1u,                                        // count=1, user mode
             lds_addr,                                  // LDS byte address
             (unsigned)(ga & 0xFFFFFFFFu),              // global addr lo
             (unsigned)((ga >> 32) & 0x01FFFFFFu)       // global addr hi (57b)
                 | (2u << 30) };                        // type = 2 ("image")
  const unsigned tw = row_stride_elems;                 // tensor_dim0 >= tile_w
  const unsigned th = tile_h;                           // tensor_dim1
  v8i g1;
  g1[0] = (int)(1u << 16);                              // data_size = 1 -> 2B
  g1[1] = (int)((tw & 0xFFFFu) << 16);                  // tensor_dim0[15:0]
  g1[2] = (int)(((tw >> 16) & 0xFFFFu)                  // tensor_dim0[31:16]
                | ((th & 0xFFFFu) << 16));              // tensor_dim1[15:0]
  g1[3] = (int)(((th >> 16) & 0xFFFFu)                  // tensor_dim1[31:16]
                | ((tile_w & 0xFFFFu) << 16));          // tile_dim0
  g1[4] = (int)(tile_h & 0xFFFFu);                      // tile_dim1, tile_dim2=0
  g1[5] = (int)row_stride_elems;                        // dim0_stride[31:0]
  g1[6] = 0;                                            // dim0_stride hi, dim1_stride lo
  g1[7] = 0;
  v4i z4 = {0,0,0,0};
  v8i z8 = {0,0,0,0,0,0,0,0};
  // 6-arg toolchain signature: (g0, g1, g2, g3, g4, cpol)
  __builtin_amdgcn_tensor_load_to_lds(g0, g1, z4, z4, z8, 0);
}

static __device__ inline unsigned lds_addr_of(const void* p) {
  return (unsigned)(uintptr_t)p;   // low 32 bits of flat addr = LDS offset
}

// ---------------------------------------------------------------------------
// RMSNorm: one token per block (D=1024, 256 threads x 4 elems), bf16 output.
// ---------------------------------------------------------------------------
__global__ __launch_bounds__(256) void rmsnorm_bf16_kernel(
    const float* __restrict__ x, const float* __restrict__ scale,
    bf16* __restrict__ out, int d) {
  const int tok = blockIdx.x;
  const int tid = threadIdx.x;
  const float4 v = ((const float4*)(x + (size_t)tok * d))[tid];
  float ss = v.x*v.x + v.y*v.y + v.z*v.z + v.w*v.w;
  #pragma unroll
  for (int m = 16; m >= 1; m >>= 1) ss += __shfl_xor(ss, m);
  __shared__ float red[8];
  if ((tid & 31) == 0) red[tid >> 5] = ss;
  __syncthreads();
  float tot = 0.f;
  #pragma unroll
  for (int i = 0; i < 8; ++i) tot += red[i];
  const float rstd = rsqrtf(tot / (float)d + 1e-6f);
  const float4 sc = ((const float4*)scale)[tid];
  v4bf o;
  o[0] = to_bf16(v.x * sc.x * rstd);
  o[1] = to_bf16(v.y * sc.y * rstd);
  o[2] = to_bf16(v.z * sc.z * rstd);
  o[3] = to_bf16(v.w * sc.w * rstd);
  ((v4bf*)(out + (size_t)tok * d))[tid] = o;
}

// ---------------------------------------------------------------------------
// Weight staging helpers (fp32 global -> regs -> bf16 LDS).
// ---------------------------------------------------------------------------
static __device__ inline void stage_w_load(const float* __restrict__ W, int nBase,
                                           int K, int k0, int tid, float4 wreg[4]) {
  #pragma unroll
  for (int i = 0; i < 4; ++i) {
    const int idx = tid + i * 128;                 // 0..511 chunks of 4 f32
    const int r = idx >> 3, kc = (idx & 7) * 4;
    wreg[i] = *(const float4*)&W[(size_t)(nBase + r) * K + k0 + kc];
  }
}

static __device__ inline void stage_w_store(bf16* Bsbuf, int tid,
                                            const float4 wreg[4]) {
  #pragma unroll
  for (int i = 0; i < 4; ++i) {
    const int idx = tid + i * 128;
    const int r = idx >> 3, kc = (idx & 7) * 4;
    v4bf b; b[0]=to_bf16(wreg[i].x); b[1]=to_bf16(wreg[i].y);
            b[2]=to_bf16(wreg[i].z); b[3]=to_bf16(wreg[i].w);
    *(v4bf*)&Bsbuf[r*32 + kc] = b;
  }
}

// ---------------------------------------------------------------------------
// GEMM: C[M,N] = A[M,K](bf16) * W[N,K](f32)^T, tile 64x64x32, 4 waves,
// wave owns 32x32 (2x2 WMMA). A tiles DMA'd by TDM; double-buffered LDS.
// EPI==0: bf16 store.  EPI==1: fp32 store with residual add.
// ---------------------------------------------------------------------------
template<int EPI>
__global__ __launch_bounds__(128) void gemm_bf16_kernel(
    const bf16* __restrict__ A, const float* __restrict__ W,
    const float* __restrict__ resid, void* __restrict__ outp,
    int M, int N, int K) {
  __shared__ __align__(32) bf16 As[2][64*32];
  __shared__ __align__(32) bf16 Bs[2][64*32];
  const int tid  = threadIdx.x;
  const int lane = tid & 31, wave = tid >> 5;
  const int half = lane >> 4, l15 = lane & 15;
  const int wm = (wave >> 1) * 32, wn = (wave & 1) * 32;
  const int mBase = blockIdx.y * 64, nBase = blockIdx.x * 64;

  v8f acc[2][2];
  #pragma unroll
  for (int mt = 0; mt < 2; ++mt)
    #pragma unroll
    for (int nt = 0; nt < 2; ++nt) acc[mt][nt] = v8f_zero();

  const int nIter = K / 32;
  // prologue: kick off tile 0 (TDM for A, VMEM for W)
  if (wave == 0)
    tdm_load_tile_bf16(&A[(size_t)mBase * K], lds_addr_of(&As[0][0]), 32, 64, K);
  float4 wreg[4];
  stage_w_load(W, nBase, K, 0, tid, wreg);
  stage_w_store(&Bs[0][0], tid, wreg);

  int cur = 0;
  for (int it = 0; it < nIter; ++it) {
    if (wave == 0) __builtin_amdgcn_s_wait_tensorcnt(0);
    __syncthreads();                    // tile[cur] visible; prev reads done
    const bool more = (it + 1 < nIter);
    if (more) {
      const int kn = (it + 1) * 32;
      if (wave == 0)
        tdm_load_tile_bf16(&A[(size_t)mBase * K + kn],
                           lds_addr_of(&As[cur ^ 1][0]), 32, 64, K);
      stage_w_load(W, nBase, K, kn, tid, wreg);   // overlaps the WMMAs below
    }

    v16bf af[2], bfr[2];
    #pragma unroll
    for (int mt = 0; mt < 2; ++mt) {
      const bf16* p = &As[cur][(wm + mt*16 + l15) * 32];
      af[mt] = *(const v16bf*)&p[half*16];
    }
    #pragma unroll
    for (int nt = 0; nt < 2; ++nt)
      bfr[nt] = *(const v16bf*)&Bs[cur][(wn + nt*16 + l15) * 32 + half*16];

    #pragma unroll
    for (int mt = 0; mt < 2; ++mt)
      #pragma unroll
      for (int nt = 0; nt < 2; ++nt)
        acc[mt][nt] = wmma_bf16(af[mt], bfr[nt], acc[mt][nt]);

    if (more) stage_w_store(&Bs[cur ^ 1][0], tid, wreg);
    cur ^= 1;
  }

  // epilogue: C layout -> row = j + 8*half, col = l15 within each 16x16 tile
  #pragma unroll
  for (int mt = 0; mt < 2; ++mt)
    #pragma unroll
    for (int nt = 0; nt < 2; ++nt)
      #pragma unroll
      for (int j = 0; j < 8; ++j) {
        const int gr = mBase + wm + mt*16 + half*8 + j;
        const int gc = nBase + wn + nt*16 + l15;
        const size_t off = (size_t)gr * N + gc;
        if (EPI == 0) {
          ((bf16*)outp)[off] = to_bf16(acc[mt][nt][j]);
        } else {
          ((float*)outp)[off] = resid[off] + acc[mt][nt][j];
        }
      }
}

// ---------------------------------------------------------------------------
// Fused SwiGLU up-projection: h = silu(A*W1^T) * (A*W2^T), bf16 out.
// Same TDM + double-buffer structure, two weight streams.
// ---------------------------------------------------------------------------
__global__ __launch_bounds__(128) void mlp_gate_kernel(
    const bf16* __restrict__ A, const float* __restrict__ W1,
    const float* __restrict__ W2, bf16* __restrict__ h,
    int M, int N, int K) {
  __shared__ __align__(32) bf16 As[2][64*32];
  __shared__ __align__(32) bf16 B1s[2][64*32];
  __shared__ __align__(32) bf16 B2s[2][64*32];
  const int tid  = threadIdx.x;
  const int lane = tid & 31, wave = tid >> 5;
  const int half = lane >> 4, l15 = lane & 15;
  const int wm = (wave >> 1) * 32, wn = (wave & 1) * 32;
  const int mBase = blockIdx.y * 64, nBase = blockIdx.x * 64;

  v8f acc1[2][2], acc2[2][2];
  #pragma unroll
  for (int mt = 0; mt < 2; ++mt)
    #pragma unroll
    for (int nt = 0; nt < 2; ++nt) { acc1[mt][nt] = v8f_zero(); acc2[mt][nt] = v8f_zero(); }

  const int nIter = K / 32;
  if (wave == 0)
    tdm_load_tile_bf16(&A[(size_t)mBase * K], lds_addr_of(&As[0][0]), 32, 64, K);
  float4 w1reg[4], w2reg[4];
  stage_w_load(W1, nBase, K, 0, tid, w1reg);
  stage_w_load(W2, nBase, K, 0, tid, w2reg);
  stage_w_store(&B1s[0][0], tid, w1reg);
  stage_w_store(&B2s[0][0], tid, w2reg);

  int cur = 0;
  for (int it = 0; it < nIter; ++it) {
    if (wave == 0) __builtin_amdgcn_s_wait_tensorcnt(0);
    __syncthreads();
    const bool more = (it + 1 < nIter);
    if (more) {
      const int kn = (it + 1) * 32;
      if (wave == 0)
        tdm_load_tile_bf16(&A[(size_t)mBase * K + kn],
                           lds_addr_of(&As[cur ^ 1][0]), 32, 64, K);
      stage_w_load(W1, nBase, K, kn, tid, w1reg);
      stage_w_load(W2, nBase, K, kn, tid, w2reg);
    }

    v16bf af[2], b1f[2], b2f[2];
    #pragma unroll
    for (int mt = 0; mt < 2; ++mt) {
      const bf16* p = &As[cur][(wm + mt*16 + l15) * 32];
      af[mt] = *(const v16bf*)&p[half*16];
    }
    #pragma unroll
    for (int nt = 0; nt < 2; ++nt) {
      b1f[nt] = *(const v16bf*)&B1s[cur][(wn + nt*16 + l15) * 32 + half*16];
      b2f[nt] = *(const v16bf*)&B2s[cur][(wn + nt*16 + l15) * 32 + half*16];
    }
    #pragma unroll
    for (int mt = 0; mt < 2; ++mt)
      #pragma unroll
      for (int nt = 0; nt < 2; ++nt) {
        acc1[mt][nt] = wmma_bf16(af[mt], b1f[nt], acc1[mt][nt]);
        acc2[mt][nt] = wmma_bf16(af[mt], b2f[nt], acc2[mt][nt]);
      }

    if (more) {
      stage_w_store(&B1s[cur ^ 1][0], tid, w1reg);
      stage_w_store(&B2s[cur ^ 1][0], tid, w2reg);
    }
    cur ^= 1;
  }

  #pragma unroll
  for (int mt = 0; mt < 2; ++mt)
    #pragma unroll
    for (int nt = 0; nt < 2; ++nt)
      #pragma unroll
      for (int j = 0; j < 8; ++j) {
        const int gr = mBase + wm + mt*16 + half*8 + j;
        const int gc = nBase + wn + nt*16 + l15;
        const float z = acc1[mt][nt][j];
        const float g = z / (1.0f + __expf(-z));       // silu
        h[(size_t)gr * N + gc] = to_bf16(g * acc2[mt][nt][j]);
      }
}

// ---------------------------------------------------------------------------
// Flash attention: block = (64-query tile, head); 4 waves x 16 queries each.
// S = Q K^T via WMMA (K-frags straight from global), online softmax with
// 16-lane shuffle reductions, P staged per-wave in LDS, O += P V with V
// transposed through LDS for contiguous B-fragments.
// ---------------------------------------------------------------------------
__global__ __launch_bounds__(128) void attn_kernel(
    const bf16* __restrict__ qkv, const int* __restrict__ mask,
    bf16* __restrict__ y) {
  __shared__ __align__(32) bf16 vT[64*64];          // V^T [hd][key] for chunk
  __shared__ __align__(32) bf16 pbuf[4][16*64];     // per-wave P tile
  const int tid  = threadIdx.x, lane = tid & 31, wave = tid >> 5;
  const int half = lane >> 4, l15 = lane & 15;
  const int qBase = blockIdx.x * 64;
  const int hOff  = blockIdx.y * HDIM;
  const int ld = 3 * DMODEL;

  // Q fragments: 16 queries x 64 hd = two 16x32 A-frags
  v16bf qf[2];
  {
    const bf16* qrow = qkv + (size_t)(qBase + wave*16 + l15) * ld + hOff;
    #pragma unroll
    for (int ks = 0; ks < 2; ++ks) {
      const int b1 = ks*32 + half*16;
      qf[ks] = *(const v16bf*)&qrow[b1];
    }
  }

  float m_i[8], l_i[8];
  #pragma unroll
  for (int j = 0; j < 8; ++j) { m_i[j] = -INFINITY; l_i[j] = 0.f; }
  v8f oc[4];
  #pragma unroll
  for (int nt = 0; nt < 4; ++nt) oc[nt] = v8f_zero();

  for (int c = 0; c < T_TOK / 64; ++c) {
    const int keyBase = c * 64;
    __syncthreads();
    // stage V chunk transposed: vT[hd][key]
    #pragma unroll
    for (int i = 0; i < 4; ++i) {
      int idx = tid + i * 128;                 // 0..511
      int kr = idx >> 3, hc = (idx & 7) * 8;
      v8bf vv = *(const v8bf*)&qkv[(size_t)(keyBase + kr) * ld
                                   + 2*DMODEL + hOff + hc];
      #pragma unroll
      for (int e = 0; e < 8; ++e) vT[(hc + e)*64 + kr] = vv[e];
    }
    __syncthreads();

    // S = Q K^T  (16 queries x 64 keys per wave)
    v8f scv[4];
    #pragma unroll
    for (int nt = 0; nt < 4; ++nt) scv[nt] = v8f_zero();
    #pragma unroll
    for (int ks = 0; ks < 2; ++ks)
      #pragma unroll
      for (int nt = 0; nt < 4; ++nt) {
        const bf16* krow = qkv + (size_t)(keyBase + nt*16 + l15) * ld
                           + DMODEL + hOff;
        v16bf kf = *(const v16bf*)&krow[ks*32 + half*16];
        scv[nt] = wmma_bf16(qf[ks], kf, scv[nt]);
      }

    // scale + key mask
    #pragma unroll
    for (int nt = 0; nt < 4; ++nt) {
      const int key = keyBase + nt*16 + l15;
      const float mk = (mask[key] != 0) ? 0.f : -INFINITY;
      #pragma unroll
      for (int j = 0; j < 8; ++j) scv[nt][j] = scv[nt][j] * 0.125f + mk;
    }

    // online softmax, per row (rows live inside one 16-lane half)
    #pragma unroll
    for (int j = 0; j < 8; ++j) {
      float mx = fmaxf(fmaxf(scv[0][j], scv[1][j]), fmaxf(scv[2][j], scv[3][j]));
      #pragma unroll
      for (int s = 8; s >= 1; s >>= 1) mx = fmaxf(mx, __shfl_xor(mx, s));
      const float mnew = fmaxf(m_i[j], mx);
      const float a = __expf(m_i[j] - mnew);
      float sum = 0.f;
      #pragma unroll
      for (int nt = 0; nt < 4; ++nt) {
        const float p = __expf(scv[nt][j] - mnew);
        scv[nt][j] = p; sum += p;
      }
      #pragma unroll
      for (int s = 8; s >= 1; s >>= 1) sum += __shfl_xor(sum, s);
      l_i[j] = l_i[j] * a + sum;
      m_i[j] = mnew;
      #pragma unroll
      for (int nt = 0; nt < 4; ++nt) oc[nt][j] *= a;
    }

    // P -> per-wave LDS tile (C layout -> [row][col] bf16)
    #pragma unroll
    for (int nt = 0; nt < 4; ++nt)
      #pragma unroll
      for (int j = 0; j < 8; ++j)
        pbuf[wave][(half*8 + j)*64 + nt*16 + l15] = to_bf16(scv[nt][j]);

    // O += P @ V
    #pragma unroll
    for (int ks = 0; ks < 2; ++ks) {
      const bf16* prow = &pbuf[wave][l15 * 64];
      const int b1 = ks*32 + half*16;
      v16bf pa = *(const v16bf*)&prow[b1];
      #pragma unroll
      for (int nt = 0; nt < 4; ++nt) {
        v16bf vb = *(const v16bf*)&vT[(nt*16 + l15)*64 + ks*32 + half*16];
        oc[nt] = wmma_bf16(pa, vb, oc[nt]);
      }
    }
  }

  // normalize and write y[t, D] (heads interleaved at hOff)
  #pragma unroll
  for (int j = 0; j < 8; ++j) {
    const float inv = 1.0f / l_i[j];
    const int qrow = qBase + wave*16 + half*8 + j;
    #pragma unroll
    for (int nt = 0; nt < 4; ++nt)
      y[(size_t)qrow * DMODEL + hOff + nt*16 + l15] = to_bf16(oc[nt][j] * inv);
  }
}

// ---------------------------------------------------------------------------
// Host launcher
// ---------------------------------------------------------------------------
extern "C" void kernel_launch(void* const* d_in, const int* in_sizes, int n_in,
                              void* d_out, int out_size, void* d_ws, size_t ws_size,
                              hipStream_t stream) {
  (void)in_sizes; (void)n_in; (void)out_size; (void)ws_size;
  const float* x    = (const float*)d_in[0];
  const int*   mask = (const int*)d_in[1];
  const float* n1   = (const float*)d_in[2];
  const float* wqkv = (const float*)d_in[3];
  const float* wo   = (const float*)d_in[4];
  const float* n2   = (const float*)d_in[5];
  const float* w1   = (const float*)d_in[6];
  const float* w2   = (const float*)d_in[7];
  const float* w3   = (const float*)d_in[8];
  float* out = (float*)d_out;

  char* ws = (char*)d_ws;
  bf16*  xn   = (bf16*) (ws);                          //  8 MB  [t, D] bf16
  bf16*  qkv  = (bf16*) (ws + ((size_t) 8 << 20));     // 24 MB  [t, 3D] bf16
  bf16*  yb   = (bf16*) (ws + ((size_t)32 << 20));     //  8 MB  [t, D] bf16
  float* x1   = (float*)(ws + ((size_t)40 << 20));     // 16 MB  [t, D] f32
  bf16*  x1n  = (bf16*) (ws + ((size_t)56 << 20));     //  8 MB  [t, D] bf16
  bf16*  hbuf = (bf16*) (ws + ((size_t)64 << 20));     // 32 MB  [t, FF] bf16

  // 1) xn = rmsnorm(x) -> bf16
  rmsnorm_bf16_kernel<<<T_TOK, 256, 0, stream>>>(x, n1, xn, DMODEL);
  // 2) qkv = xn @ w_qkv^T -> bf16 [t, 3D]
  gemm_bf16_kernel<0><<<dim3(3*DMODEL/64, T_TOK/64), 128, 0, stream>>>(
      xn, wqkv, nullptr, qkv, T_TOK, 3*DMODEL, DMODEL);
  // 3) flash attention -> y bf16 [t, D]
  attn_kernel<<<dim3(T_TOK/64, NHEAD), 128, 0, stream>>>(qkv, mask, yb);
  // 4) x1 = x + y @ w_o^T -> f32
  gemm_bf16_kernel<1><<<dim3(DMODEL/64, T_TOK/64), 128, 0, stream>>>(
      yb, wo, x, x1, T_TOK, DMODEL, DMODEL);
  // 5) x1n = rmsnorm(x1) -> bf16
  rmsnorm_bf16_kernel<<<T_TOK, 256, 0, stream>>>(x1, n2, x1n, DMODEL);
  // 6) h = silu(x1n @ w1^T) * (x1n @ w2^T) -> bf16 [t, FF]
  mlp_gate_kernel<<<dim3(FFDIM/64, T_TOK/64), 128, 0, stream>>>(
      x1n, w1, w2, hbuf, T_TOK, FFDIM, DMODEL);
  // 7) out = x1 + h @ w3^T -> f32 [t, D]
  gemm_bf16_kernel<1><<<dim3(DMODEL/64, T_TOK/64), 128, 0, stream>>>(
      hbuf, w3, x1, out, T_TOK, DMODEL, FFDIM);
}
